// PromptedSwinTransformerBlock_12850542149943
// MI455X (gfx1250) — compile-verified, matrix operand order checked by reading, not run on hardware
//
#include <hip/hip_runtime.h>
#include <math.h>

// ---------------------------------------------------------------------------
// Prompted Swin block for MI455X (gfx1250, wave32).
// GEMMs: v_wmma_f32_16x16x32_bf16 with fp32 accumulation.
// Tile staging: global_load_async_to_lds_b128 (ASYNCcnt) + double buffering.
// ---------------------------------------------------------------------------

typedef __attribute__((ext_vector_type(16))) __bf16 v16bf;
typedef __attribute__((ext_vector_type(4)))  __bf16 v4bf;
typedef __attribute__((ext_vector_type(2)))  __bf16 v2bf;
typedef __attribute__((ext_vector_type(8)))  float  v8f;

#define B_SZ   64
#define C_DIM  512
#define H_DIM  28
#define W_DIM  28
#define WS_    7
#define SH_    3
#define NH_    16
#define P_TOK  10
#define MLP_HD 2048
#define NW_    16
#define N_TOK  59            // P + 7*7
#define NPAD   64
#define HEAD_D 32
#define NWIN   (B_SZ * NW_)  // 1024 windows
#define HW_    (H_DIM * W_DIM)
#define XROW   (P_TOK + HW_) // 794 tokens per batch row in input x
#define M_ATT  (NWIN * NPAD) // 65536 padded attention rows
#define M_TOK  (B_SZ * HW_)  // 50176 image tokens
#define SCALE_ 0.1767766953f // 32^-0.5

#define REGION(h) ((h) < (H_DIM - WS_) ? 0 : ((h) < (H_DIM - SH_) ? 1 : 2))

// ---- CDNA5 async global->LDS copy (16B per lane, tracked by ASYNCcnt) -----
__device__ inline unsigned lds_off(const void* p) {
  // generic pointer to LDS: addr[31:0] is the LDS byte offset
  return (unsigned)(unsigned long long)p;
}
__device__ inline void async_copy_b128(unsigned lds, const void* g) {
  unsigned long long ga = (unsigned long long)g;
  asm volatile("global_load_async_to_lds_b128 %0, %1, off"
               :: "v"(lds), "v"(ga)
               : "memory");
}
__device__ inline void wait_async_le2() {
  asm volatile("s_wait_asynccnt 2" ::: "memory");
}
__device__ inline void wait_async_0() {
  asm volatile("s_wait_asynccnt 0" ::: "memory");
}

// ---- WMMA fragment helpers (16-bit A-matrix 16x32 layout, ISA 7.12.2) -----
__device__ inline v16bf load_frag(const __bf16* base, int ld) {
  int lane = threadIdx.x & 31;
  int r    = lane & 15;
  int kb   = (lane & 16) ? 8 : 0;
  const __bf16* p = base + r * ld;
  v16bf f;
#pragma unroll
  for (int i = 0; i < 8; ++i) f[i] = p[kb + i];
#pragma unroll
  for (int i = 0; i < 8; ++i) f[8 + i] = p[16 + kb + i];
  return f;
}

// ---- shared GEMM mainloop: 64x64 tile, 8 waves, double-buffered async LDS -
// sA/sB must each hold 2*64*32 bf16 (two 4KB buffers).
template <int K>
__device__ inline void gemm_tile(const __bf16* __restrict__ A,
                                 const __bf16* __restrict__ W,
                                 int m0, int n0,
                                 __bf16* sA, __bf16* sB,
                                 v8f& acc0, v8f& acc1) {
  const int KC = K / 32;
  int tid = threadIdx.x;
  int wv  = tid >> 5;
  int r   = tid >> 2;         // 0..63
  int c   = (tid & 3) * 8;    // 0,8,16,24
  unsigned la = lds_off(sA + tid * 8);
  unsigned lb = lds_off(sB + tid * 8);
  const __bf16* gA = A + (size_t)(m0 + r) * K + c;
  const __bf16* gW = W + (size_t)(n0 + r) * K + c;

  // prologue: chunk 0 -> buffer 0
  async_copy_b128(la, gA);
  async_copy_b128(lb, gW);

  for (int k = 0; k < KC; ++k) {
    int buf = k & 1;
    if (k + 1 < KC) {  // prefetch chunk k+1 into the other buffer
      int nb = buf ^ 1;
      async_copy_b128(la + nb * 4096, gA + (k + 1) * 32);
      async_copy_b128(lb + nb * 4096, gW + (k + 1) * 32);
      wait_async_le2();   // chunk k resident (async loads complete in order)
    } else {
      wait_async_0();
    }
    __syncthreads();
    const __bf16* bA = sA + buf * 2048;
    const __bf16* bB = sB + buf * 2048;
#pragma unroll
    for (int s = 0; s < 2; ++s) {
      int id = wv * 2 + s;
      int rt = id >> 2, ct = id & 3;
      v16bf fa = load_frag(bA + rt * 16 * 32, 32);
      v16bf fb = load_frag(bB + ct * 16 * 32, 32);
      v8f& acc = s ? acc1 : acc0;
      acc = __builtin_amdgcn_wmma_f32_16x16x32_bf16(false, fa, false, fb,
                                                    (short)0, acc, false, false);
    }
    __syncthreads();  // readers done before buffer is overwritten next iter
  }
}

// ---- LN statistics over 512 elements, 256 threads, 2 elems/thread --------
__device__ inline void ln_stats(float a0, float a1, float* red, int tid,
                                float& mu, float& rstd) {
  red[tid] = a0 + a1;
  __syncthreads();
  for (int off = 128; off > 0; off >>= 1) {
    if (tid < off) red[tid] += red[tid + off];
    __syncthreads();
  }
  mu = red[0] * (1.0f / C_DIM);
  __syncthreads();
  float d0 = a0 - mu, d1 = a1 - mu;
  red[tid] = d0 * d0 + d1 * d1;
  __syncthreads();
  for (int off = 128; off > 0; off >>= 1) {
    if (tid < off) red[tid] += red[tid + off];
    __syncthreads();
  }
  rstd = rsqrtf(red[0] * (1.0f / C_DIM) + 1e-5f);
  __syncthreads();
}

// ---------------------------------------------------------------------------
__global__ void cast_f32_bf16(const float* __restrict__ src,
                              __bf16* __restrict__ dst, int n4) {
  int i = blockIdx.x * blockDim.x + threadIdx.x;
  if (i < n4) {
    float4 f = ((const float4*)src)[i];
    v4bf o;
    o[0] = (__bf16)f.x; o[1] = (__bf16)f.y;
    o[2] = (__bf16)f.z; o[3] = (__bf16)f.w;
    ((v4bf*)dst)[i] = o;
  }
}

// LN1 + roll(-SH,-SH) + window partition + prompt broadcast -> xw bf16
__global__ void ln1_window(const float* __restrict__ x,
                           const float* __restrict__ g,
                           const float* __restrict__ bta,
                           __bf16* __restrict__ xw) {
  __shared__ float red[256];
  int blk = blockIdx.x;
  int b_  = blk >> 6;
  int t   = blk & 63;
  int tid = threadIdx.x;
  __bf16* dst = xw + ((size_t)b_ * NPAD + t) * C_DIM;
  if (t >= N_TOK) {  // pad rows
    v2bf z; z[0] = (__bf16)0.0f; z[1] = (__bf16)0.0f;
    *(v2bf*)(dst + 2 * tid) = z;
    return;
  }
  int b = b_ >> 4;
  size_t orow;
  bool donorm;
  if (t < P_TOK) {  // prompt tokens: raw (no LN)
    orow = (size_t)b * XROW + t;
    donorm = false;
  } else {
    int wi = b_ & 15;
    int u = t - P_TOK;
    int i = u / WS_, j = u % WS_;
    int hs  = (wi >> 2) * WS_ + i;
    int wsc = (wi & 3) * WS_ + j;
    int h = (hs + SH_) % H_DIM, w = (wsc + SH_) % W_DIM;
    orow = (size_t)b * XROW + P_TOK + h * W_DIM + w;
    donorm = true;
  }
  float2 a = *(const float2*)(x + orow * C_DIM + 2 * tid);
  v2bf o;
  if (donorm) {
    float mu, rstd;
    ln_stats(a.x, a.y, red, tid, mu, rstd);
    float2 gg = *(const float2*)(g + 2 * tid);
    float2 bb = *(const float2*)(bta + 2 * tid);
    o[0] = (__bf16)((a.x - mu) * rstd * gg.x + bb.x);
    o[1] = (__bf16)((a.y - mu) * rstd * gg.y + bb.y);
  } else {
    o[0] = (__bf16)a.x;
    o[1] = (__bf16)a.y;
  }
  *(v2bf*)(dst + 2 * tid) = o;
}

// Fused per-(window,head) attention: QKV GEMM -> S=QK^T -> softmax -> P*V
__global__ void attn_fused(const __bf16* __restrict__ xw,
                           const __bf16* __restrict__ wqkv,
                           const float* __restrict__ qkv_b,
                           const float* __restrict__ rpb,
                           __bf16* __restrict__ ctx) {
  __shared__ __bf16 sA[64 * 32];   // xw K-chunk
  __shared__ __bf16 sW[96 * 32];   // q/k/v weight rows for this head
  __shared__ __bf16 sQ[64 * 32];
  __shared__ __bf16 sK[64 * 32];
  __shared__ __bf16 sVT[32 * 64];  // V transposed (d-major)
  __shared__ float  sS[64 * 64];
  __shared__ __bf16 sP[64 * 64];

  int b_   = blockIdx.x >> 4;
  int head = blockIdx.x & 15;
  int tid  = threadIdx.x;
  int wv   = tid >> 5;
  int lane = tid & 31;
  int nn   = lane & 15;
  int mo   = (lane & 16) ? 8 : 0;

  const __bf16* Ab = xw + (size_t)b_ * NPAD * C_DIM;

  // ---- QKV: (64 x 512) x (512 x 96), async-staged K chunks ----
  v8f accq[3] = {};
  {
    int rA = tid >> 2;
    int cA = (tid & 3) * 8;
    unsigned la = lds_off(sA + tid * 8);
    // weight tile: rows 0..95 of {q,k,v} slices for this head
    int e1 = tid * 8;                       // first 2048 elements
    int e2 = tid * 8 + 2048;                // remaining 1024 (tid < 128)
    int r1 = e1 >> 5, c1 = e1 & 31;
    int r2 = e2 >> 5, c2 = e2 & 31;
    auto wrow = [&](int r) {
      return (r < 32) ? (head * HEAD_D + r)
           : (r < 64) ? (C_DIM + head * HEAD_D + (r - 32))
                      : (2 * C_DIM + head * HEAD_D + (r - 64));
    };
    int wr1 = wrow(r1);
    int wr2 = (e2 < 96 * 32) ? wrow(r2) : 0;
    unsigned lw1 = lds_off(sW + e1);
    unsigned lw2 = lds_off(sW + (e2 < 96 * 32 ? e2 : 0));

    for (int k0 = 0; k0 < C_DIM; k0 += 32) {
      async_copy_b128(la, Ab + (size_t)rA * C_DIM + k0 + cA);
      async_copy_b128(lw1, wqkv + (size_t)wr1 * C_DIM + k0 + c1);
      if (e2 < 96 * 32)
        async_copy_b128(lw2, wqkv + (size_t)wr2 * C_DIM + k0 + c2);
      wait_async_0();
      __syncthreads();
#pragma unroll
      for (int s = 0; s < 3; ++s) {
        int id = wv * 3 + s;
        int rt = id / 6, ct = id % 6;
        v16bf fa = load_frag(sA + rt * 16 * 32, 32);
        v16bf fb = load_frag(sW + ct * 16 * 32, 32);
        accq[s] = __builtin_amdgcn_wmma_f32_16x16x32_bf16(false, fa, false, fb,
                                                          (short)0, accq[s],
                                                          false, false);
      }
      __syncthreads();
    }
  }
  // scatter to Q (scaled+biased) / K / V^T
#pragma unroll
  for (int s = 0; s < 3; ++s) {
    int id = wv * 3 + s;
    int rt = id / 6, ct = id % 6;
#pragma unroll
    for (int v = 0; v < 8; ++v) {
      int row = rt * 16 + mo + v;
      int col = ct * 16 + nn;
      float val = accq[s][v];
      if (col < 32) {
        sQ[row * 32 + col] =
            (__bf16)((val + qkv_b[head * HEAD_D + col]) * SCALE_);
      } else if (col < 64) {
        int c2 = col - 32;
        sK[row * 32 + c2] = (__bf16)(val + qkv_b[C_DIM + head * HEAD_D + c2]);
      } else {
        int c2 = col - 64;
        sVT[c2 * 64 + row] =
            (__bf16)(val + qkv_b[2 * C_DIM + head * HEAD_D + c2]);
      }
    }
  }
  __syncthreads();

  // ---- S = Q K^T (reduction over head_dim=32: one WMMA per tile) ----
  int wi = b_ & 15;
  int wy = wi >> 2, wx = wi & 3;
#pragma unroll
  for (int s = 0; s < 2; ++s) {
    int id = wv * 2 + s;
    int rt = id >> 2, ctm = id & 3;
    v16bf fq = load_frag(sQ + rt * 16 * 32, 32);
    v16bf fk = load_frag(sK + ctm * 16 * 32, 32);
    v8f accs = {};
    accs = __builtin_amdgcn_wmma_f32_16x16x32_bf16(false, fq, false, fk,
                                                   (short)0, accs, false, false);
#pragma unroll
    for (int v = 0; v < 8; ++v) {
      int t1 = rt * 16 + mo + v;
      int t2 = ctm * 16 + nn;
      float val = accs[v];
      if (t2 >= N_TOK) {
        val = -1e9f;  // padded key columns
      } else if (t1 >= P_TOK && t1 < N_TOK && t2 >= P_TOK) {
        int u1 = t1 - P_TOK, u2 = t2 - P_TOK;
        int i1 = u1 / WS_, j1 = u1 % WS_;
        int i2 = u2 / WS_, j2 = u2 % WS_;
        int rpi = (i1 - i2 + WS_ - 1) * (2 * WS_ - 1) + (j1 - j2 + WS_ - 1);
        val += rpb[rpi * NH_ + head];
        int r1 = REGION(wy * WS_ + i1) * 3 + REGION(wx * WS_ + j1);
        int r2 = REGION(wy * WS_ + i2) * 3 + REGION(wx * WS_ + j2);
        if (r1 != r2) val -= 100.0f;
      }
      sS[t1 * 64 + t2] = val;
    }
  }
  __syncthreads();

  // ---- softmax rows ----
  if (tid < 64) {
    float mx = -1e30f;
    for (int c = 0; c < 64; ++c) mx = fmaxf(mx, sS[tid * 64 + c]);
    float sum = 0.f;
    for (int c = 0; c < 64; ++c) {
      float e = __expf(sS[tid * 64 + c] - mx);
      sS[tid * 64 + c] = e;
      sum += e;
    }
    float inv = 1.0f / sum;
    for (int c = 0; c < 64; ++c)
      sP[tid * 64 + c] = (__bf16)(sS[tid * 64 + c] * inv);
  }
  __syncthreads();

  // ---- ctx = P * V (reduction over 64 tokens: 2 WMMA steps) ----
  {
    int rt = wv >> 1, ctd = wv & 1;
    v8f acc = {};
#pragma unroll
    for (int ks = 0; ks < 2; ++ks) {
      v16bf fp = load_frag(sP + rt * 16 * 64 + ks * 32, 64);
      v16bf fv = load_frag(sVT + ctd * 16 * 64 + ks * 32, 64);
      acc = __builtin_amdgcn_wmma_f32_16x16x32_bf16(false, fp, false, fv,
                                                    (short)0, acc, false, false);
    }
#pragma unroll
    for (int v = 0; v < 8; ++v) {
      int t = rt * 16 + mo + v;
      int d = ctd * 16 + nn;
      float val = (t < N_TOK) ? acc[v] : 0.0f;
      ctx[((size_t)b_ * NPAD + t) * C_DIM + head * HEAD_D + d] = (__bf16)val;
    }
  }
}

// proj GEMM + window_reverse + roll(+SH,+SH) + shortcut add -> res1 fp32
__global__ void proj_gemm(const __bf16* __restrict__ A,
                          const __bf16* __restrict__ W,
                          const float* __restrict__ bias,
                          const float* __restrict__ xin,
                          float* __restrict__ res1) {
  __shared__ __bf16 sA[2 * 64 * 32];
  __shared__ __bf16 sB[2 * 64 * 32];
  int m0 = blockIdx.x * 64, n0 = blockIdx.y * 64;
  v8f acc0 = {}, acc1 = {};
  gemm_tile<C_DIM>(A, W, m0, n0, sA, sB, acc0, acc1);
  int tid = threadIdx.x, wv = tid >> 5, lane = tid & 31;
  int nn = lane & 15, mo = (lane & 16) ? 8 : 0;
#pragma unroll
  for (int s = 0; s < 2; ++s) {
    v8f acc = s ? acc1 : acc0;
    int id = wv * 2 + s;
    int rt = id >> 2, ct = id & 3;
#pragma unroll
    for (int v = 0; v < 8; ++v) {
      int r = m0 + rt * 16 + mo + v;
      int n = n0 + ct * 16 + nn;
      int b_ = r >> 6, t = r & 63;
      if (t >= P_TOK && t < N_TOK) {
        int u = t - P_TOK;
        int i = u / WS_, j = u % WS_;
        int wi = b_ & 15, b = b_ >> 4;
        int hs  = (wi >> 2) * WS_ + i;
        int wsc = (wi & 3) * WS_ + j;
        int h = (hs + SH_) % H_DIM, w = (wsc + SH_) % W_DIM;
        size_t orow = (size_t)b * HW_ + h * W_DIM + w;
        res1[orow * C_DIM + n] =
            xin[((size_t)b * XROW + P_TOK + h * W_DIM + w) * C_DIM + n] +
            acc[v] + bias[n];
      }
    }
  }
}

__global__ void ln2_kernel(const float* __restrict__ res1,
                           const float* __restrict__ g,
                           const float* __restrict__ bta,
                           __bf16* __restrict__ xn2) {
  __shared__ float red[256];
  int row = blockIdx.x, tid = threadIdx.x;
  float2 a = *(const float2*)(res1 + (size_t)row * C_DIM + 2 * tid);
  float mu, rstd;
  ln_stats(a.x, a.y, red, tid, mu, rstd);
  float2 gg = *(const float2*)(g + 2 * tid);
  float2 bb = *(const float2*)(bta + 2 * tid);
  v2bf o;
  o[0] = (__bf16)((a.x - mu) * rstd * gg.x + bb.x);
  o[1] = (__bf16)((a.y - mu) * rstd * gg.y + bb.y);
  *(v2bf*)(xn2 + (size_t)row * C_DIM + 2 * tid) = o;
}

// fc1 GEMM + exact GELU -> h1 bf16
__global__ void fc1_gemm(const __bf16* __restrict__ A,
                         const __bf16* __restrict__ W,
                         const float* __restrict__ bias,
                         __bf16* __restrict__ out) {
  __shared__ __bf16 sA[2 * 64 * 32];
  __shared__ __bf16 sB[2 * 64 * 32];
  int m0 = blockIdx.x * 64, n0 = blockIdx.y * 64;
  v8f acc0 = {}, acc1 = {};
  gemm_tile<C_DIM>(A, W, m0, n0, sA, sB, acc0, acc1);
  int tid = threadIdx.x, wv = tid >> 5, lane = tid & 31;
  int nn = lane & 15, mo = (lane & 16) ? 8 : 0;
#pragma unroll
  for (int s = 0; s < 2; ++s) {
    v8f acc = s ? acc1 : acc0;
    int id = wv * 2 + s;
    int rt = id >> 2, ct = id & 3;
#pragma unroll
    for (int v = 0; v < 8; ++v) {
      int r = m0 + rt * 16 + mo + v;
      int n = n0 + ct * 16 + nn;
      float val = acc[v] + bias[n];
      float gl = 0.5f * val * (1.0f + erff(val * 0.70710678f));
      out[(size_t)r * MLP_HD + n] = (__bf16)gl;
    }
  }
}

// fc2 GEMM + residual -> d_out fp32
__global__ void fc2_gemm(const __bf16* __restrict__ A,
                         const __bf16* __restrict__ W,
                         const float* __restrict__ bias,
                         const float* __restrict__ res1,
                         float* __restrict__ out) {
  __shared__ __bf16 sA[2 * 64 * 32];
  __shared__ __bf16 sB[2 * 64 * 32];
  int m0 = blockIdx.x * 64, n0 = blockIdx.y * 64;
  v8f acc0 = {}, acc1 = {};
  gemm_tile<MLP_HD>(A, W, m0, n0, sA, sB, acc0, acc1);
  int tid = threadIdx.x, wv = tid >> 5, lane = tid & 31;
  int nn = lane & 15, mo = (lane & 16) ? 8 : 0;
#pragma unroll
  for (int s = 0; s < 2; ++s) {
    v8f acc = s ? acc1 : acc0;
    int id = wv * 2 + s;
    int rt = id >> 2, ct = id & 3;
#pragma unroll
    for (int v = 0; v < 8; ++v) {
      int r = m0 + rt * 16 + mo + v;
      int n = n0 + ct * 16 + nn;
      out[(size_t)r * C_DIM + n] =
          res1[(size_t)r * C_DIM + n] + acc[v] + bias[n];
    }
  }
}

// ---------------------------------------------------------------------------
extern "C" void kernel_launch(void* const* d_in, const int* in_sizes, int n_in,
                              void* d_out, int out_size, void* d_ws,
                              size_t ws_size, hipStream_t stream) {
  const float* x      = (const float*)d_in[0];
  const float* g1     = (const float*)d_in[1];
  const float* b1     = (const float*)d_in[2];
  const float* qkv_w  = (const float*)d_in[3];
  const float* qkv_b  = (const float*)d_in[4];
  const float* rpb    = (const float*)d_in[5];
  const float* proj_w = (const float*)d_in[6];
  const float* proj_b = (const float*)d_in[7];
  const float* g2     = (const float*)d_in[8];
  const float* b2     = (const float*)d_in[9];
  const float* fc1_w  = (const float*)d_in[10];
  const float* fc1_b  = (const float*)d_in[11];
  const float* fc2_w  = (const float*)d_in[12];
  const float* fc2_b  = (const float*)d_in[13];
  float* out = (float*)d_out;

  char* ws = (char*)d_ws;
  size_t off = 0;
  auto alloc = [&](size_t bytes) -> void* {
    void* p = ws + off;
    off = (off + bytes + 255) & ~(size_t)255;
    return p;
  };

  __bf16* wqkv_bf  = (__bf16*)alloc((size_t)3 * C_DIM * C_DIM * 2);
  __bf16* wproj_bf = (__bf16*)alloc((size_t)C_DIM * C_DIM * 2);
  __bf16* wfc1_bf  = (__bf16*)alloc((size_t)MLP_HD * C_DIM * 2);
  __bf16* wfc2_bf  = (__bf16*)alloc((size_t)C_DIM * MLP_HD * 2);
  __bf16* xw   = (__bf16*)alloc((size_t)M_ATT * C_DIM * 2);
  __bf16* ctx  = (__bf16*)alloc((size_t)M_ATT * C_DIM * 2);
  float*  res1 = (float*)alloc((size_t)M_TOK * C_DIM * 4);
  __bf16* xn2  = (__bf16*)alloc((size_t)M_TOK * C_DIM * 2);
  __bf16* h1   = (__bf16*)alloc((size_t)M_TOK * MLP_HD * 2);

  int n4;
  n4 = 3 * C_DIM * C_DIM / 4;
  cast_f32_bf16<<<(n4 + 255) / 256, 256, 0, stream>>>(qkv_w, wqkv_bf, n4);
  n4 = C_DIM * C_DIM / 4;
  cast_f32_bf16<<<(n4 + 255) / 256, 256, 0, stream>>>(proj_w, wproj_bf, n4);
  n4 = MLP_HD * C_DIM / 4;
  cast_f32_bf16<<<(n4 + 255) / 256, 256, 0, stream>>>(fc1_w, wfc1_bf, n4);
  n4 = C_DIM * MLP_HD / 4;
  cast_f32_bf16<<<(n4 + 255) / 256, 256, 0, stream>>>(fc2_w, wfc2_bf, n4);

  ln1_window<<<M_ATT, 256, 0, stream>>>(x, g1, b1, xw);

  attn_fused<<<NWIN * NH_, 256, 0, stream>>>(xw, wqkv_bf, qkv_b, rpb, ctx);

  proj_gemm<<<dim3(M_ATT / 64, C_DIM / 64), 256, 0, stream>>>(
      ctx, wproj_bf, proj_b, x, res1);

  ln2_kernel<<<M_TOK, 256, 0, stream>>>(res1, g2, b2, xn2);

  fc1_gemm<<<dim3(M_TOK / 64, MLP_HD / 64), 256, 0, stream>>>(
      xn2, wfc1_bf, fc1_b, h1);

  fc2_gemm<<<dim3(M_TOK / 64, C_DIM / 64), 256, 0, stream>>>(
      h1, wfc2_bf, fc2_b, res1, out);
}
